// Flux2Attention_30193620091126
// MI455X (gfx1250) — matrix operand order repaired and checked by reading, not compiled, not run on hardware
//
#include <hip/hip_runtime.h>
#include <hip/hip_bf16.h>

typedef __attribute__((ext_vector_type(8)))  float  v8f;
typedef __attribute__((ext_vector_type(8)))  __bf16 v8bf;
typedef __attribute__((ext_vector_type(16))) __bf16 v16bf;

#define DM   3072
#define ST   4608
#define STXT 512
#define SIMG 4096
#define NH   24
#define HD   128
#define ATTN_SCALE 0.08838834764831845f

// LDS pitches (elements) chosen for 16B alignment + bank-conflict-free B-fragment reads
#define KPITCH 136   // 272 B/row: 68 banks -> 16 rows cover distinct banks
#define VPITCH 40    // 80 B/row:  20 banks -> 16 rows cover distinct banks

static __device__ __forceinline__ v8f wmma_bf16(v16bf a, v16bf b, v8f c) {
    // D = A(16x32 bf16) * B(32x16 bf16) + C(16x16 f32)
    return __builtin_amdgcn_wmma_f32_16x16x32_bf16(false, a, false, b, (short)0, c, false, false);
}
static __device__ __forceinline__ v16bf cat8(v8bf lo, v8bf hi) {
    return __builtin_shufflevector(lo, hi, 0,1,2,3,4,5,6,7,8,9,10,11,12,13,14,15);
}

// ---- CDNA5 async global->LDS copy (ASYNCcnt path), ISA 10.1/15.18.3 --------
static __device__ __forceinline__ unsigned lds_addr32(const void* p) {
    // generic LDS pointer: low 32 bits are the LDS byte address (aperture scheme)
    return (unsigned)(unsigned long long)p;
}
static __device__ __forceinline__ void async_copy_b128(unsigned lds_off, const void* gptr) {
    // VDST = LDS byte address (u32 VGPR), VADDR = 64-bit global address, no SADDR
    asm volatile("global_load_async_to_lds_b128 %0, %1, off"
                 :: "v"(lds_off), "v"(gptr)
                 : "memory");
}
#define ASYNC_WAIT(N) asm volatile("s_wait_asynccnt %0" :: "n"(N) : "memory")

// ---------------------------------------------------------------------------
// f32 -> bf16 conversion
// ---------------------------------------------------------------------------
__global__ void cvt_bf16_kernel(const float* __restrict__ in, __bf16* __restrict__ out, long n) {
    long i = (long)blockIdx.x * blockDim.x + threadIdx.x;
    if (i < n) out[i] = (__bf16)in[i];
}

// ---------------------------------------------------------------------------
// GEMM: C[M,N] = A[M,K] * W[N,K]^T (+bias), bf16 in / f32 out.
// Block = 256 threads = 8 waves; tile 128x128; wave w -> rows (w/2)*32, cols (w&1)*64.
// A fragment (16x32 bf16): lane L row = L%16, K = (L/16)*16 + {0..15}.
// B fragment (32x16 bf16): lane L col = L%16, K = (L/16)*16 + {0..15} (contiguous in W row).
// ---------------------------------------------------------------------------
__global__ __launch_bounds__(256)
void gemm_bf16_kernel(const __bf16* __restrict__ A, const __bf16* __restrict__ W,
                      const float* __restrict__ bias, float* __restrict__ C) {
    const int K = DM;
    const int lane = threadIdx.x & 31;
    const int wave = threadIdx.x >> 5;
    const int lrow = lane & 15;
    const int lhi  = lane >> 4;
    const int rowBase = blockIdx.x * 128 + (wave >> 1) * 32;
    const int colBase = blockIdx.y * 128 + (wave & 1) * 64;

    v8f acc[2][4] = {};
    for (int k0 = 0; k0 < K; k0 += 32) {
        v16bf a[2], b[4];
#pragma unroll
        for (int rt = 0; rt < 2; ++rt) {
            const __bf16* ap = A + (size_t)(rowBase + rt * 16 + lrow) * K + k0 + lhi * 16;
            a[rt] = cat8(*(const v8bf*)ap, *(const v8bf*)(ap + 8));
        }
#pragma unroll
        for (int nt = 0; nt < 4; ++nt) {
            const __bf16* bp = W + (size_t)(colBase + nt * 16 + lrow) * K + k0 + lhi * 16;
            b[nt] = *(const v16bf*)bp;
        }
#pragma unroll
        for (int rt = 0; rt < 2; ++rt)
#pragma unroll
            for (int nt = 0; nt < 4; ++nt)
                acc[rt][nt] = wmma_bf16(a[rt], b[nt], acc[rt][nt]);
    }
#pragma unroll
    for (int rt = 0; rt < 2; ++rt)
#pragma unroll
        for (int nt = 0; nt < 4; ++nt) {
            int col = colBase + nt * 16 + lrow;
            float bv = bias ? bias[col] : 0.f;
#pragma unroll
            for (int r = 0; r < 8; ++r) {
                int row = rowBase + rt * 16 + r + 8 * lhi;
                C[(size_t)row * DM + col] = acc[rt][nt][r] + bv;
            }
        }
}

// ---------------------------------------------------------------------------
// RMSNorm + RoPE epilogue for Q/K. One wave per (s,h); lane handles 4 dims.
// Output layout: [H][S_total][HD] bf16.
// ---------------------------------------------------------------------------
__global__ __launch_bounds__(256)
void prep_qk_kernel(const float* __restrict__ Sf, const float* __restrict__ wtxt,
                    const float* __restrict__ wimg, const float* __restrict__ rc,
                    const float* __restrict__ rsn, __bf16* __restrict__ out) {
    const int wave = threadIdx.x >> 5;
    const int lane = threadIdx.x & 31;
    const int pair = blockIdx.x * 8 + wave;
    const int s = pair / NH;
    const int h = pair % NH;
    const int d0 = lane * 4;

    const float* xp = Sf + (size_t)s * DM + h * HD + d0;
    float x0 = xp[0], x1 = xp[1], x2 = xp[2], x3 = xp[3];
    float ss = x0 * x0 + x1 * x1 + x2 * x2 + x3 * x3;
    ss += __shfl_xor(ss, 1, 32);
    ss += __shfl_xor(ss, 2, 32);
    ss += __shfl_xor(ss, 4, 32);
    ss += __shfl_xor(ss, 8, 32);
    ss += __shfl_xor(ss, 16, 32);
    float rstd = rsqrtf(ss * (1.0f / HD) + 1e-5f);
    const float* wn = (s < STXT) ? wtxt : wimg;
    x0 *= rstd * wn[d0 + 0]; x1 *= rstd * wn[d0 + 1];
    x2 *= rstd * wn[d0 + 2]; x3 *= rstd * wn[d0 + 3];

    const float* cp = rc  + (size_t)s * HD + d0;
    const float* sp = rsn + (size_t)s * HD + d0;
    float o0 = x0 * cp[0] - x1 * sp[0];
    float o1 = x1 * cp[1] + x0 * sp[1];
    float o2 = x2 * cp[2] - x3 * sp[2];
    float o3 = x3 * cp[3] + x2 * sp[3];

    __bf16* op = out + ((size_t)h * ST + s) * HD + d0;
    op[0] = (__bf16)o0; op[1] = (__bf16)o1; op[2] = (__bf16)o2; op[3] = (__bf16)o3;
}

// ---------------------------------------------------------------------------
// V epilogue: transpose into [H][HD][S_total] bf16 so PV B-fragments are contiguous.
// ---------------------------------------------------------------------------
__global__ void prep_v_kernel(const float* __restrict__ Sf, __bf16* __restrict__ Vt) {
    size_t i = (size_t)blockIdx.x * blockDim.x + threadIdx.x;
    const size_t total = (size_t)NH * HD * ST;
    if (i >= total) return;
    int s = (int)(i % ST);
    int d = (int)((i / ST) % HD);
    int h = (int)(i / ((size_t)ST * HD));
    Vt[i] = (__bf16)Sf[(size_t)s * DM + h * HD + d];
}

// ---------------------------------------------------------------------------
// Flash attention. Block = 4 waves / 64 queries; wave w owns queries q0+16w.
// K (32x128) and V^T (128x32) chunks are double-buffered in LDS via
// global_load_async_to_lds_b128 (ASYNCcnt) and shared by all 4 waves.
// Per chunk/wave: 8 WMMAs QK^T + online softmax + P via LDS + 8 WMMAs PV.
// ---------------------------------------------------------------------------
__global__ __launch_bounds__(128)
void attn_kernel(const __bf16* __restrict__ Qb, const __bf16* __restrict__ Kb,
                 const __bf16* __restrict__ Vt, __bf16* __restrict__ AO) {
    const int tid  = threadIdx.x;
    const int lane = tid & 31;
    const int wv   = tid >> 5;          // 0..3
    const int lrow = lane & 15;
    const int lhi  = lane >> 4;
    const int h  = blockIdx.y;
    const int q0 = blockIdx.x * 64 + wv * 16;

    const __bf16* Qh = Qb + (size_t)h * ST * HD;
    const __bf16* Kh = Kb + (size_t)h * ST * HD;
    const __bf16* Vh = Vt + (size_t)h * HD * ST;

    __shared__ __align__(16) __bf16 kt[2][32 * KPITCH];
    __shared__ __align__(16) __bf16 vt[2][128 * VPITCH];
    __shared__ __align__(16) __bf16 pls[4][16 * 32];

    // per-lane async copy assignment
    const int cseg = (lane & 3) * 8;             // element offset of this lane's 16B
    const int krow = wv * 8 + (lane >> 2);       // K row owned by this lane (0..31)

    // Q fragments (A-layout), resident in VGPRs for the whole pass
    v16bf qf[4];
#pragma unroll
    for (int c = 0; c < 4; ++c) {
        const __bf16* qp = Qh + (size_t)(q0 + lrow) * HD + c * 32 + lhi * 16;
        qf[c] = cat8(*(const v8bf*)qp, *(const v8bf*)(qp + 8));
    }

    v8f acc[8] = {};
    float m[8], l[8];
#pragma unroll
    for (int r = 0; r < 8; ++r) { m[r] = -1e30f; l[r] = 0.f; }

    auto issue_chunk = [&](int buf, int kc) {
        // K chunk: rows kc..kc+31, contiguous 256B rows. Each wave copies 8 rows.
#pragma unroll
        for (int i = 0; i < 4; ++i) {
            const __bf16* g = Kh + (size_t)(kc + krow) * HD + i * 32 + cseg;
            async_copy_b128(lds_addr32(&kt[buf][krow * KPITCH + i * 32 + cseg]), g);
        }
        // V^T chunk: 128 rows x 32 keys (64B rows, global stride ST*2B).
#pragma unroll
        for (int i = 0; i < 4; ++i) {
            int vrow = wv * 32 + i * 8 + (lane >> 2);
            const __bf16* g = Vh + (size_t)vrow * ST + kc + cseg;
            async_copy_b128(lds_addr32(&vt[buf][vrow * VPITCH + cseg]), g);
        }
    };

    const int NCH = ST / 32;          // 144 chunks
    issue_chunk(0, 0);

    for (int i = 0; i < NCH; ++i) {
        const int buf = i & 1;
        if (i + 1 < NCH) {
            issue_chunk(buf ^ 1, (i + 1) * 32);
            ASYNC_WAIT(8);            // in-order: first 8 (chunk i) have landed
        } else {
            ASYNC_WAIT(0);
        }
        __syncthreads();              // all waves' copies of chunk i visible

        // --- QK^T: B-fragments from LDS K tile -----------------------------
        v8f s0 = {}, s1 = {};
#pragma unroll
        for (int c = 0; c < 4; ++c) {
            const __bf16* kp0 = &kt[buf][(lrow)      * KPITCH + c * 32 + lhi * 16];
            const __bf16* kp1 = &kt[buf][(16 + lrow) * KPITCH + c * 32 + lhi * 16];
            s0 = wmma_bf16(qf[c], *(const v16bf*)kp0, s0);
            s1 = wmma_bf16(qf[c], *(const v16bf*)kp1, s1);
        }
        // --- online softmax (rows live in 16-lane halves) ------------------
        float p0[8], p1[8], alpha[8];
#pragma unroll
        for (int r = 0; r < 8; ++r) {
            float a = s0[r] * ATTN_SCALE;
            float b = s1[r] * ATTN_SCALE;
            float rm = fmaxf(a, b);
            rm = fmaxf(rm, __shfl_xor(rm, 1, 32));
            rm = fmaxf(rm, __shfl_xor(rm, 2, 32));
            rm = fmaxf(rm, __shfl_xor(rm, 4, 32));
            rm = fmaxf(rm, __shfl_xor(rm, 8, 32));
            float mn = fmaxf(m[r], rm);
            alpha[r] = __expf(m[r] - mn);
            float e0 = __expf(a - mn);
            float e1 = __expf(b - mn);
            float rs = e0 + e1;
            rs += __shfl_xor(rs, 1, 32);
            rs += __shfl_xor(rs, 2, 32);
            rs += __shfl_xor(rs, 4, 32);
            rs += __shfl_xor(rs, 8, 32);
            l[r] = l[r] * alpha[r] + rs;
            m[r] = mn;
            p0[r] = e0; p1[r] = e1;
        }
#pragma unroll
        for (int n = 0; n < 8; ++n)
#pragma unroll
            for (int r = 0; r < 8; ++r)
                acc[n][r] *= alpha[r];

        // --- P: C-layout -> LDS -> A-layout fragment (per-wave region) -----
#pragma unroll
        for (int r = 0; r < 8; ++r) {
            int prow = r + 8 * lhi;
            pls[wv][prow * 32 + lrow]      = (__bf16)p0[r];
            pls[wv][prow * 32 + 16 + lrow] = (__bf16)p1[r];
        }
        const __bf16* pp = &pls[wv][lrow * 32 + lhi * 16];
        v16bf pf = cat8(*(const v8bf*)pp, *(const v8bf*)(pp + 8));

        // --- PV: B-fragments from LDS V^T tile -----------------------------
#pragma unroll
        for (int n = 0; n < 8; ++n) {
            const __bf16* vp = &vt[buf][(n * 16 + lrow) * VPITCH + lhi * 16];
            acc[n] = wmma_bf16(pf, *(const v16bf*)vp, acc[n]);
        }
        __syncthreads();              // done reading buf before it is rewritten
    }

#pragma unroll
    for (int n = 0; n < 8; ++n) {
        int col = h * HD + n * 16 + lrow;
#pragma unroll
        for (int r = 0; r < 8; ++r) {
            int row = q0 + r + 8 * lhi;
            AO[(size_t)row * DM + col] = (__bf16)(acc[n][r] / l[r]);
        }
    }
}

// ---------------------------------------------------------------------------
// Host orchestration
// ---------------------------------------------------------------------------
extern "C" void kernel_launch(void* const* d_in, const int* in_sizes, int n_in,
                              void* d_out, int out_size, void* d_ws, size_t ws_size,
                              hipStream_t stream) {
    (void)in_sizes; (void)n_in; (void)out_size; (void)ws_size;
    const float* hs   = (const float*)d_in[0];
    const float* enc  = (const float*)d_in[1];
    const float* rc   = (const float*)d_in[2];
    const float* rsn  = (const float*)d_in[3];
    const float* Wq   = (const float*)d_in[4];
    const float* Wk   = (const float*)d_in[5];
    const float* Wv   = (const float*)d_in[6];
    const float* Waq  = (const float*)d_in[7];
    const float* baq  = (const float*)d_in[8];
    const float* Wak  = (const float*)d_in[9];
    const float* bak  = (const float*)d_in[10];
    const float* Wav  = (const float*)d_in[11];
    const float* bav  = (const float*)d_in[12];
    const float* Wo   = (const float*)d_in[13];
    const float* bo   = (const float*)d_in[14];
    const float* Wao  = (const float*)d_in[15];
    const float* bao  = (const float*)d_in[16];
    const float* wqn  = (const float*)d_in[17];
    const float* wkn  = (const float*)d_in[18];
    const float* waqn = (const float*)d_in[19];
    const float* wakn = (const float*)d_in[20];

    char* base = (char*)d_ws;
    size_t off = 0;
    auto take = [&](size_t bytes) -> void* {
        void* p = (void*)(base + off);
        off += (bytes + 255) & ~(size_t)255;
        return p;
    };
    __bf16* hs_bf  = (__bf16*)take((size_t)SIMG * DM * 2);
    __bf16* enc_bf = (__bf16*)take((size_t)STXT * DM * 2);
    __bf16* wbf[8];
    for (int i = 0; i < 8; ++i) wbf[i] = (__bf16*)take((size_t)DM * DM * 2);
    float*  Sf   = (float*) take((size_t)ST * DM * 4);   // reused f32 scratch for Q, K, V
    __bf16* Qb   = (__bf16*)take((size_t)NH * ST * HD * 2);
    __bf16* Kb   = (__bf16*)take((size_t)NH * ST * HD * 2);
    __bf16* Vt   = (__bf16*)take((size_t)NH * ST * HD * 2);
    __bf16* AObf = (__bf16*)take((size_t)ST * DM * 2);

    auto cvt = [&](const float* in, __bf16* out, long n) {
        int blk = (int)((n + 255) / 256);
        cvt_bf16_kernel<<<dim3(blk), dim3(256), 0, stream>>>(in, out, n);
    };
    cvt(hs, hs_bf, (long)SIMG * DM);
    cvt(enc, enc_bf, (long)STXT * DM);
    const float* wsrc[8] = {Wq, Wk, Wv, Waq, Wak, Wav, Wo, Wao};
    for (int i = 0; i < 8; ++i) cvt(wsrc[i], wbf[i], (long)DM * DM);

    auto gemm = [&](const __bf16* A, const __bf16* W, const float* bias, float* C, int M) {
        gemm_bf16_kernel<<<dim3(M / 128, DM / 128), dim3(256), 0, stream>>>(A, W, bias, C);
    };

    // Q: text rows [0,512) then image rows [512,4608) of the joint sequence
    gemm(enc_bf, wbf[3], baq, Sf, STXT);
    gemm(hs_bf, wbf[0], nullptr, Sf + (size_t)STXT * DM, SIMG);
    prep_qk_kernel<<<dim3(ST * NH / 8), dim3(256), 0, stream>>>(Sf, waqn, wqn, rc, rsn, Qb);
    // K
    gemm(enc_bf, wbf[4], bak, Sf, STXT);
    gemm(hs_bf, wbf[1], nullptr, Sf + (size_t)STXT * DM, SIMG);
    prep_qk_kernel<<<dim3(ST * NH / 8), dim3(256), 0, stream>>>(Sf, wakn, wkn, rc, rsn, Kb);
    // V (no norm / rope) -> transposed V^T
    gemm(enc_bf, wbf[5], bav, Sf, STXT);
    gemm(hs_bf, wbf[2], nullptr, Sf + (size_t)STXT * DM, SIMG);
    {
        long total = (long)NH * HD * ST;
        prep_v_kernel<<<dim3((int)((total + 255) / 256)), dim3(256), 0, stream>>>(Sf, Vt);
    }
    // attention over joint sequence
    attn_kernel<<<dim3(ST / 64, NH), dim3(128), 0, stream>>>(Qb, Kb, Vt, AObf);
    // output projections: d_out = [img_out (4096x3072), enc_out (512x3072)]
    float* out_img = (float*)d_out;
    float* out_enc = (float*)d_out + (size_t)SIMG * DM;
    gemm(AObf + (size_t)STXT * DM, wbf[6], bo, out_img, SIMG);
    gemm(AObf, wbf[7], bao, out_enc, STXT);
}
